// MixerBlock_66374424592558
// MI455X (gfx1250) — matrix operand order repaired and checked by reading, not compiled, not run on hardware
//
#include <hip/hip_runtime.h>
#include <hip/hip_bf16.h>
#include <stdint.h>

// ---------------------------------------------------------------------------
// MixerBlock for MI455X (gfx1250): bf16 WMMA (v_wmma_f32_16x16x32_bf16) GEMMs
// ---------------------------------------------------------------------------

typedef __attribute__((ext_vector_type(16))) __bf16 v16bf;
typedef __attribute__((ext_vector_type(8)))  float  v8f;

#define BB 4096
#define NJ 17
#define DD 512
#define CC 1024
#define MROWS (BB * NJ)   // 69632
#define LDT 40            // padded LDS tile leading dim (half-words) to dodge bank conflicts

// ---------------- scalar helpers ----------------
__device__ __forceinline__ unsigned short f2bf(float f) {
    unsigned int u = __float_as_uint(f);
    unsigned int r = u + 0x7FFFu + ((u >> 16) & 1u);
    return (unsigned short)(r >> 16);
}
__device__ __forceinline__ float bf2f(unsigned short s) {
    return __uint_as_float(((unsigned int)s) << 16);
}
// Branchless tanh-form gelu: one v_exp_f32 + few FMAs (no saveexec divergence).
__device__ __forceinline__ float geluf(float v) {
    float u = v * (0.7978845608028654f + 0.03567740814183870f * v * v);
    float a = __expf(-2.0f * fabsf(u));       // (0,1]
    float t = (1.0f - a) / (1.0f + a);        // tanh(|u|)
    t = copysignf(t, u);
    return 0.5f * v * (1.0f + t);
}
__device__ __forceinline__ v8f v8zero() {
    v8f z;
#pragma unroll
    for (int i = 0; i < 8; i++) z[i] = 0.0f;
    return z;
}

union FragU {
    v16bf v;
    unsigned short e[16];
    uint4 q[2];
};

// A-fragment (16x32 bf16, M x K) from row-major (K-contiguous) tile.
// p must point at  tile + (lane&15)*ld + ((lane>=16)?8:0)   (ushort elements)
__device__ __forceinline__ v16bf load_a_frag(const unsigned short* p) {
    FragU u;
    u.q[0] = *(const uint4*)(p);        // K = kb .. kb+7
    u.q[1] = *(const uint4*)(p + 16);   // K = kb+16 .. kb+23
    return u.v;
}
// B-fragment (32x16 bf16, K x N) from K-contiguous (BT) tile.
// p must point at  tileT + n*ld + ((lane>=16)?8:0)
__device__ __forceinline__ v16bf load_b_frag(const unsigned short* p) {
    FragU u;
    u.q[0] = *(const uint4*)(p);        // K = koff .. koff+7
    u.q[1] = *(const uint4*)(p + 16);   // K = koff+16 .. koff+23
    return u.v;
}

// ---------------- K0a: transpose W1/W2 to bf16, K-contiguous ----------------
__global__ void prep_wT_kernel(const float* __restrict__ W1, const float* __restrict__ W2,
                               unsigned short* __restrict__ W1T, unsigned short* __restrict__ W2T) {
    int idx = blockIdx.x * blockDim.x + threadIdx.x;      // 0 .. 524287
    int c = idx >> 9, d = idx & 511;                      // W1T[c][d] = W1[d][c]
    W1T[idx] = f2bf(W1[d * CC + c]);
    int d2 = idx >> 10, c2 = idx & 1023;                  // W2T[d2][c2] = W2[c2][d2]
    W2T[idx] = f2bf(W2[c2 * DD + d2]);
}

// ---------------- K0b: W12T[dout][din] = bf16(alpha2 * sum_c W1[din][c]*W2[c][dout]) ----
__global__ void prep_w12_kernel(const float* __restrict__ W1, const float* __restrict__ W2,
                                const float* __restrict__ alpha2p, unsigned short* __restrict__ W12T) {
    int din = blockIdx.x;
    float a2 = alpha2p[0];
    for (int dout = threadIdx.x; dout < DD; dout += blockDim.x) {
        float s = 0.0f;
        for (int c = 0; c < CC; c++) s += W1[din * CC + c] * W2[c * DD + dout];
        W12T[dout * DD + din] = f2bf(a2 * s);
    }
}

// ---------------- K0c: fc1T [512][32], fc2T [32][512], adj_eff ----------------
__global__ void prep_small_kernel(const float* __restrict__ fc1_w, const float* __restrict__ fc2_w,
                                  const float* __restrict__ adj,
                                  const float* __restrict__ adjmod1, const float* __restrict__ adjmod2,
                                  unsigned short* __restrict__ fc1T, unsigned short* __restrict__ fc2T,
                                  float* __restrict__ adjE1, float* __restrict__ adjE2) {
    int idx = blockIdx.x * blockDim.x + threadIdx.x;
    if (idx < DD * 32) {  // fc1T[o][k] = fc1_w[k][o], k padded 17->32 with zeros
        int o = idx >> 5, k = idx & 31;
        fc1T[idx] = (k < NJ) ? f2bf(fc1_w[k * DD + o]) : (unsigned short)0;
    }
    if (idx < 32 * DD) {  // fc2T[n][k] = fc2_w[k][n], n padded 17->32 with zeros
        int n = idx >> 9, k = idx & 511;
        fc2T[idx] = (n < NJ) ? f2bf(fc2_w[k * NJ + n]) : (unsigned short)0;
    }
    if (idx < NJ * NJ) {
        adjE1[idx] = adj[idx] + adjmod1[idx];
        adjE2[idx] = adj[idx] + adjmod2[idx];
    }
}

// ---------------- K1: LayerNorm row statistics (mean, rstd) ----------------
__global__ __launch_bounds__(256) void ln_stats_kernel(const float* __restrict__ x,
                                                       float* __restrict__ lnstats) {
    int row = blockIdx.x;         // b*17 + n
    int tid = threadIdx.x;
    float v0 = x[(size_t)row * DD + tid];
    float v1 = x[(size_t)row * DD + tid + 256];
    float s = v0 + v1, q = v0 * v0 + v1 * v1;
    __shared__ float sS[256], sQ[256];
    sS[tid] = s; sQ[tid] = q;
    __syncthreads();
    for (int off = 128; off > 0; off >>= 1) {
        if (tid < off) { sS[tid] += sS[tid + off]; sQ[tid] += sQ[tid + off]; }
        __syncthreads();
    }
    if (tid == 0) {
        float m = sS[0] * (1.0f / DD);
        float var = sQ[0] * (1.0f / DD) - m * m;
        lnstats[row * 2] = m;
        lnstats[row * 2 + 1] = rsqrtf(var + 1e-5f);
    }
}

// ---------------- K2: fused spatial MLP (LN -> fc1 -> gelu -> fc2 -> residual) ----
// grid = (DD/64, BB), block = 128 (4 waves); wave handles 16 rows (d values)
__global__ __launch_bounds__(128) void spatial_mlp_kernel(
    const float* __restrict__ x, const float* __restrict__ lnstats,
    const float* __restrict__ ln_g, const float* __restrict__ ln_b,
    const unsigned short* __restrict__ fc1T, const float* __restrict__ fc1_b,
    const unsigned short* __restrict__ fc2T, const float* __restrict__ fc2_b,
    float* __restrict__ xr)
{
    __shared__ unsigned short stage[4][16 * LDT];
    __shared__ float lst[2 * NJ];
    const int tid = threadIdx.x;
    const int wave = tid >> 5, lane = tid & 31;
    const int half = lane >> 4, l15 = lane & 15;
    const int b = blockIdx.y;
    const int m0 = blockIdx.x * 64;
    if (tid < 2 * NJ) lst[tid] = lnstats[b * (2 * NJ) + tid];
    __syncthreads();

    const int d = m0 + wave * 16 + l15;     // this lane's row (A matrix M index)
    const float lg = ln_g[d], lb = ln_b[d];

    // Build A1 fragment on the fly: A1[d][k] = LN(x)[b][k][d], k = joint (0-pad to 32)
    FragU a1;
#pragma unroll
    for (int e = 0; e < 16; e++) {
        int k = (e & 7) + ((e >> 3) << 4) + half * 8;  // ISA A-layout K index
        unsigned short val = 0;
        if (k < NJ) {
            float xv = x[((size_t)b * NJ + k) * DD + d];
            val = f2bf((xv - lst[k * 2]) * lst[k * 2 + 1] * lg + lb);
        }
        a1.e[e] = val;
    }

    v8f acc2[2];
    acc2[0] = v8zero(); acc2[1] = v8zero();

    for (int j = 0; j < DD; j += 32) {
        // GEMM1: C1[16x32] = A1[16x32(joints)] @ fc1_w[32 x cols j..j+31]
#pragma unroll
        for (int h = 0; h < 2; h++) {
            int col = j + h * 16 + l15;
            v16bf b1 = load_b_frag(&fc1T[col * 32 + half * 8]);
            v8f c1 = v8zero();
            c1 = __builtin_amdgcn_wmma_f32_16x16x32_bf16(false, a1.v, false, b1,
                                                         (short)0, c1, false, false);
            float bias = fc1_b[col];
#pragma unroll
            for (int r = 0; r < 8; r++) {
                float v = geluf(c1[r] + bias);
                // C layout: m = r + 8*half, n = l15 (+ h*16 within chunk)
                stage[wave][(r + half * 8) * LDT + h * 16 + l15] = f2bf(v);
            }
        }
        // A2 fragment (accumulator-layout -> A-layout via LDS)
        v16bf a2 = load_a_frag(&stage[wave][l15 * LDT + half * 8]);
        // GEMM2 accumulate: acc2 += A2[16x32] @ fc2_w[j..j+31, 0..31(pad)]
#pragma unroll
        for (int t = 0; t < 2; t++) {
            v16bf b2 = load_b_frag(&fc2T[(t * 16 + l15) * DD + j + half * 8]);
            acc2[t] = __builtin_amdgcn_wmma_f32_16x16x32_bf16(false, a2, false, b2,
                                                              (short)0, acc2[t], false, false);
        }
    }

    // Epilogue: xr[b][n][d'] = x[b][n][d'] + acc2 + fc2_b[n]  (n < 17)
#pragma unroll
    for (int t = 0; t < 2; t++) {
        int n = t * 16 + l15;
        if (n < NJ) {
            float bias = fc2_b[n];
            size_t base = ((size_t)b * NJ + n) * DD + m0 + wave * 16 + half * 8;
#pragma unroll
            for (int r = 0; r < 8; r++)
                xr[base + r] = x[base + r] + acc2[t][r] + bias;
        }
    }
}

// ---------------- K3: support1 = (1-a1)*AdjE1.xr + a1*initial  (+ bf16(initial)) ----
__global__ __launch_bounds__(256) void support1_kernel(
    const float* __restrict__ xr, const float* __restrict__ initial,
    const float* __restrict__ adjE1, const float* __restrict__ alpha1p,
    unsigned short* __restrict__ s1b, unsigned short* __restrict__ initb)
{
    __shared__ float adjs[NJ * NJ];
    int tid = threadIdx.x, b = blockIdx.x;
    for (int i = tid; i < NJ * NJ; i += blockDim.x) adjs[i] = adjE1[i];
    __syncthreads();
    float a1 = alpha1p[0];
    for (int dd = tid; dd < DD; dd += blockDim.x) {
        float xv[NJ];
        size_t base = (size_t)b * NJ * DD + dd;
#pragma unroll
        for (int jj = 0; jj < NJ; jj++) xv[jj] = xr[base + jj * DD];
        for (int ii = 0; ii < NJ; ii++) {
            float agg = 0.0f;
#pragma unroll
            for (int jj = 0; jj < NJ; jj++) agg += adjs[ii * NJ + jj] * xv[jj];
            float iv = initial[base + ii * DD];
            s1b[base + ii * DD]  = f2bf((1.0f - a1) * agg + a1 * iv);
            initb[base + ii * DD] = f2bf(iv);
        }
    }
}

// ---------------- K7: s2b = bf16((1-a2) * AdjE2 . gelu(BN1(g1)))  (BN1+gelu fused) ----
__global__ __launch_bounds__(256) void support2_kernel(
    const unsigned short* __restrict__ g1b, const float* __restrict__ adjE2,
    const float* __restrict__ alpha2p,
    const float* __restrict__ ch1s, const float* __restrict__ ch1b,
    unsigned short* __restrict__ s2b)
{
    __shared__ float adjs[NJ * NJ];
    int tid = threadIdx.x, b = blockIdx.x;
    for (int i = tid; i < NJ * NJ; i += blockDim.x) adjs[i] = adjE2[i];
    __syncthreads();
    float w = 1.0f - alpha2p[0];
    for (int c = tid; c < CC; c += blockDim.x) {
        float sc = ch1s[c], sh = ch1b[c];
        float hv[NJ];
        size_t base = (size_t)b * NJ * CC + c;
#pragma unroll
        for (int jj = 0; jj < NJ; jj++)
            hv[jj] = geluf(bf2f(g1b[base + jj * CC]) * sc + sh);   // h1 = gelu(BN1(g1))
        for (int ii = 0; ii < NJ; ii++) {
            float agg = 0.0f;
#pragma unroll
            for (int jj = 0; jj < NJ; jj++) agg += adjs[ii * NJ + jj] * hv[jj];
            s2b[base + ii * CC] = f2bf(w * agg);
        }
    }
}

// ---------------- WMMA GEMM: Out[M,N] (+)= A[M,K] @ B (BT is [N,K] bf16) ----------
// 128x64 block tile, BK=32; 4 waves; wave: 32 rows x 64 cols (8 WMMA / k-step,
// each B-fragment reused by two A-fragments). LDS padded to LDT to avoid bank
// conflicts. Next A k-tile prefetched (global_prefetch_b8).
__global__ __launch_bounds__(128) void wmma_gemm_kernel(
    const unsigned short* __restrict__ A, const unsigned short* __restrict__ BT,
    float* __restrict__ OutF, unsigned short* __restrict__ OutH,
    int M, int N, int K, int accumulate)
{
    __shared__ unsigned short As[128 * LDT];
    __shared__ unsigned short Bs[64 * LDT];
    const int tid = threadIdx.x;
    const int wave = tid >> 5, lane = tid & 31;
    const int half = lane >> 4, l15 = lane & 15;
    const size_t row0 = (size_t)blockIdx.y * 128;
    const size_t col0 = (size_t)blockIdx.x * 64;

    v8f acc[2][4];
#pragma unroll
    for (int s = 0; s < 2; s++)
#pragma unroll
        for (int t = 0; t < 4; t++) acc[s][t] = v8zero();

    for (int k0 = 0; k0 < K; k0 += 32) {
#pragma unroll
        for (int i = 0; i < 4; i++) {          // A: 128 rows x 64B = 512 x uint4
            int idx = tid + i * 128;
            int r = idx >> 2, c4 = idx & 3;
            *(uint4*)&As[r * LDT + c4 * 8] =
                *(const uint4*)&A[(row0 + r) * (size_t)K + k0 + c4 * 8];
        }
#pragma unroll
        for (int i = 0; i < 2; i++) {          // B: 64 rows x 64B = 256 x uint4
            int idx = tid + i * 128;
            int r = idx >> 2, c4 = idx & 3;
            *(uint4*)&Bs[r * LDT + c4 * 8] =
                *(const uint4*)&BT[(col0 + r) * (size_t)K + k0 + c4 * 8];
        }
        if (k0 + 32 < K)                       // prefetch next A k-tile (one line/row)
            __builtin_prefetch(&A[(row0 + tid) * (size_t)K + k0 + 32], 0, 1);
        __syncthreads();

        v16bf af0 = load_a_frag(&As[(wave * 32 + l15) * LDT + half * 8]);
        v16bf af1 = load_a_frag(&As[(wave * 32 + 16 + l15) * LDT + half * 8]);
#pragma unroll
        for (int t = 0; t < 4; t++) {
            v16bf bf = load_b_frag(&Bs[(t * 16 + l15) * LDT + half * 8]);
            acc[0][t] = __builtin_amdgcn_wmma_f32_16x16x32_bf16(false, af0, false, bf,
                                                                (short)0, acc[0][t], false, false);
            acc[1][t] = __builtin_amdgcn_wmma_f32_16x16x32_bf16(false, af1, false, bf,
                                                                (short)0, acc[1][t], false, false);
        }
        __syncthreads();
    }

#pragma unroll
    for (int s = 0; s < 2; s++) {
        const size_t mrow = row0 + wave * 32 + s * 16 + half * 8;
#pragma unroll
        for (int t = 0; t < 4; t++) {
            size_t col = col0 + t * 16 + l15;
#pragma unroll
            for (int r = 0; r < 8; r++) {
                size_t o = (mrow + r) * (size_t)N + col;
                float v = acc[s][t][r];
                if (OutF) {
                    if (accumulate) v += OutF[o];
                    OutF[o] = v;
                } else {
                    OutH[o] = f2bf(v);
                }
            }
        }
    }
}

// ---------------- column statistics (two-stage, deterministic) ----------------
__global__ void col_stats_kernel(const void* __restrict__ X, int isBf16, int M, int N,
                                 float* __restrict__ partS, float* __restrict__ partQ) {
    // grid: (N/64, NPART), block (64,4)
    int c = blockIdx.x * 64 + threadIdx.x;
    int stride = gridDim.y * 4;
    float s = 0.0f, q = 0.0f;
    for (int m = blockIdx.y * 4 + threadIdx.y; m < M; m += stride) {
        float v = isBf16 ? bf2f(((const unsigned short*)X)[(size_t)m * N + c])
                         : ((const float*)X)[(size_t)m * N + c];
        s += v; q += v * v;
    }
    __shared__ float shS[4][64], shQ[4][64];
    shS[threadIdx.y][threadIdx.x] = s;
    shQ[threadIdx.y][threadIdx.x] = q;
    __syncthreads();
    if (threadIdx.y == 0) {
#pragma unroll
        for (int i = 1; i < 4; i++) { s += shS[i][threadIdx.x]; q += shQ[i][threadIdx.x]; }
        partS[blockIdx.y * N + c] = s;
        partQ[blockIdx.y * N + c] = q;
    }
}

__global__ void finalize_stats_kernel(const float* __restrict__ partS, const float* __restrict__ partQ,
                                      int N, int nparts, int M,
                                      const float* __restrict__ g, const float* __restrict__ b,
                                      float* __restrict__ scale, float* __restrict__ shift) {
    int c = blockIdx.x * blockDim.x + threadIdx.x;
    if (c >= N) return;
    float s = 0.0f, q = 0.0f;
    for (int p = 0; p < nparts; p++) { s += partS[p * N + c]; q += partQ[p * N + c]; }
    float mean = s / (float)M;
    float var = q / (float)M - mean * mean;
    float sc = g[c] * rsqrtf(var + 1e-5f);
    scale[c] = sc;
    shift[c] = b[c] - mean * sc;
}

// ---------------- final: out = xr + gelu(BN2(g2))  (float4 / thread-iter) ----
__global__ void final_out_kernel(const float* __restrict__ xr, const float* __restrict__ g2,
                                 const float* __restrict__ scale, const float* __restrict__ shift,
                                 float* __restrict__ out, size_t totalQuads) {
    size_t stride = (size_t)gridDim.x * blockDim.x;
    for (size_t qd = (size_t)blockIdx.x * blockDim.x + threadIdx.x; qd < totalQuads; qd += stride) {
        size_t i = qd * 4;
        int c = (int)(i & (size_t)(DD - 1));
        float4 g = *(const float4*)&g2[i];
        float4 xv = *(const float4*)&xr[i];
        float4 r;
        r.x = xv.x + geluf(g.x * scale[c]     + shift[c]);
        r.y = xv.y + geluf(g.y * scale[c + 1] + shift[c + 1]);
        r.z = xv.z + geluf(g.z * scale[c + 2] + shift[c + 2]);
        r.w = xv.w + geluf(g.w * scale[c + 3] + shift[c + 3]);
        *(float4*)&out[i] = r;
    }
}

// ===========================================================================
extern "C" void kernel_launch(void* const* d_in, const int* in_sizes, int n_in,
                              void* d_out, int out_size, void* d_ws, size_t ws_size,
                              hipStream_t stream) {
    (void)in_sizes; (void)n_in; (void)out_size; (void)ws_size;

    const float* x       = (const float*)d_in[0];
    const float* initial = (const float*)d_in[1];
    const float* adj     = (const float*)d_in[2];
    const float* ln_g    = (const float*)d_in[3];
    const float* ln_b    = (const float*)d_in[4];
    const float* fc1_w   = (const float*)d_in[5];
    const float* fc1_b   = (const float*)d_in[6];
    const float* fc2_w   = (const float*)d_in[7];
    const float* fc2_b   = (const float*)d_in[8];
    const float* W1      = (const float*)d_in[9];
    const float* adjmod1 = (const float*)d_in[10];
    const float* alpha1  = (const float*)d_in[11];
    const float* bn1_g   = (const float*)d_in[12];
    const float* bn1_b   = (const float*)d_in[13];
    const float* W2      = (const float*)d_in[14];
    const float* adjmod2 = (const float*)d_in[15];
    const float* alpha2  = (const float*)d_in[16];
    const float* bn2_g   = (const float*)d_in[17];
    const float* bn2_b   = (const float*)d_in[18];
    float* out = (float*)d_out;

    char* ws = (char*)d_ws;
    const size_t SZ_XR   = (size_t)MROWS * DD * 4;   // 142,606,336
    const size_t SZ_G2   = (size_t)MROWS * DD * 4;   // front half doubles as s1b
    const size_t SZ_INIT = (size_t)MROWS * DD * 2;
    const size_t SZ_G1B  = (size_t)MROWS * CC * 2;
    const size_t SZ_S2B  = (size_t)MROWS * CC * 2;

    size_t off = 0;
    auto take = [&](size_t n) { size_t r = off; off += (n + 255) & ~(size_t)255; return r; };
    const size_t o_xr    = take(SZ_XR);
    const size_t o_g2    = take(SZ_G2);
    const size_t o_initb = take(SZ_INIT);
    const size_t o_g1b   = take(SZ_G1B);
    const size_t o_s2b   = take(SZ_S2B);
    const size_t o_w1t   = take((size_t)DD * CC * 2);
    const size_t o_w2t   = take((size_t)CC * DD * 2);
    const size_t o_w12t  = take((size_t)DD * DD * 2);
    const size_t o_fc1t  = take((size_t)DD * 32 * 2);
    const size_t o_fc2t  = take((size_t)32 * DD * 2);
    const size_t o_adje1 = take(NJ * NJ * 4);
    const size_t o_adje2 = take(NJ * NJ * 4);
    const size_t o_lnst  = take((size_t)MROWS * 2 * 4);
    const size_t o_p1s   = take(32 * CC * 4);
    const size_t o_p1q   = take(32 * CC * 4);
    const size_t o_ch1s  = take(CC * 4);
    const size_t o_ch1b  = take(CC * 4);
    const size_t o_p2s   = take(32 * DD * 4);
    const size_t o_p2q   = take(32 * DD * 4);
    const size_t o_ch2s  = take(DD * 4);
    const size_t o_ch2b  = take(DD * 4);

    float*          xr    = (float*)(ws + o_xr);
    float*          g2    = (float*)(ws + o_g2);
    unsigned short* s1b   = (unsigned short*)(ws + o_g2);   // alias; dead before g2 written
    unsigned short* initb = (unsigned short*)(ws + o_initb);
    unsigned short* g1b   = (unsigned short*)(ws + o_g1b);
    unsigned short* s2b   = (unsigned short*)(ws + o_s2b);
    unsigned short* W1T   = (unsigned short*)(ws + o_w1t);
    unsigned short* W2T   = (unsigned short*)(ws + o_w2t);
    unsigned short* W12T  = (unsigned short*)(ws + o_w12t);
    unsigned short* fc1T  = (unsigned short*)(ws + o_fc1t);
    unsigned short* fc2T  = (unsigned short*)(ws + o_fc2t);
    float*          adjE1 = (float*)(ws + o_adje1);
    float*          adjE2 = (float*)(ws + o_adje2);
    float*          lnst  = (float*)(ws + o_lnst);
    float*          p1s   = (float*)(ws + o_p1s);
    float*          p1q   = (float*)(ws + o_p1q);
    float*          ch1s  = (float*)(ws + o_ch1s);
    float*          ch1b  = (float*)(ws + o_ch1b);
    float*          p2s   = (float*)(ws + o_p2s);
    float*          p2q   = (float*)(ws + o_p2q);
    float*          ch2s  = (float*)(ws + o_ch2s);
    float*          ch2b  = (float*)(ws + o_ch2b);

    // ---- weight preprocessing ----
    prep_wT_kernel<<<2048, 256, 0, stream>>>(W1, W2, W1T, W2T);
    prep_w12_kernel<<<512, 256, 0, stream>>>(W1, W2, alpha2, W12T);
    prep_small_kernel<<<64, 256, 0, stream>>>(fc1_w, fc2_w, adj, adjmod1, adjmod2,
                                              fc1T, fc2T, adjE1, adjE2);

    // ---- spatial mix ----
    ln_stats_kernel<<<MROWS, 256, 0, stream>>>(x, lnst);
    spatial_mlp_kernel<<<dim3(DD / 64, BB), 128, 0, stream>>>(
        x, lnst, ln_g, ln_b, fc1T, fc1_b, fc2T, fc2_b, xr);

    // ---- graph conv 1 ----
    support1_kernel<<<BB, 256, 0, stream>>>(xr, initial, adjE1, alpha1, s1b, initb);
    wmma_gemm_kernel<<<dim3(CC / 64, MROWS / 128), 128, 0, stream>>>(
        s1b, W1T, nullptr, g1b, MROWS, CC, DD, 0);
    col_stats_kernel<<<dim3(CC / 64, 32), dim3(64, 4), 0, stream>>>(g1b, 1, MROWS, CC, p1s, p1q);
    finalize_stats_kernel<<<4, 256, 0, stream>>>(p1s, p1q, CC, 32, MROWS, bn1_g, bn1_b, ch1s, ch1b);

    // ---- graph conv 2 (BN1+gelu fused into the aggregation; init2@W2 folded
    //      into initb @ (alpha2*W1W2)) ----
    support2_kernel<<<BB, 256, 0, stream>>>(g1b, adjE2, alpha2, ch1s, ch1b, s2b);
    wmma_gemm_kernel<<<dim3(DD / 64, MROWS / 128), 128, 0, stream>>>(
        s2b, W2T, g2, nullptr, MROWS, DD, CC, 0);
    wmma_gemm_kernel<<<dim3(DD / 64, MROWS / 128), 128, 0, stream>>>(
        initb, W12T, g2, nullptr, MROWS, DD, DD, 1);
    col_stats_kernel<<<dim3(DD / 64, 32), dim3(64, 4), 0, stream>>>(g2, 0, MROWS, DD, p2s, p2q);
    finalize_stats_kernel<<<2, 256, 0, stream>>>(p2s, p2q, DD, 32, MROWS, bn2_g, bn2_b, ch2s, ch2b);

    // ---- output ----
    final_out_kernel<<<8192, 256, 0, stream>>>(xr, g2, ch2s, ch2b, out, (size_t)MROWS * DD / 4);
}
